// CCAM_Module_75359496175786
// MI455X (gfx1250) — compile-verified, matrix operand order checked by reading, not run on hardware
//
#include <hip/hip_runtime.h>
#include <hip/hip_bf16.h>

// MI455X / gfx1250, wave32. Fused channel-attention (CCAM):
//   energy = x(16384x4096) @ martx(4096x64)      [bf16 WMMA, f32 acc]
//   att    = softmax(aphal * (rowmax - energy))  [cross-lane shfl reduction]
//   out    = gamma * (att @ martx^T) + x         [bf16 WMMA, f32 acc, fused residual]

typedef __attribute__((ext_vector_type(16))) __bf16 v16bf;
typedef __attribute__((ext_vector_type(8)))  float  v8f;

union BF16x16 {
  unsigned u[8];
  v16bf    v;
};

#define NDIM    4096
#define KD      64
#define WAVES   8
#define BLOCKT  256
#define NCHUNK  32
#define NITERS  (NDIM / NCHUNK)   // 128
#define BSTRIDE 40                // bf16 elems per staged martx column (32 + 8 pad, 80B: 16B aligned, bank-spread)
#define ASTRIDE 72                // bf16 elems per attention row     (64 + 8 pad, 144B: 16B aligned, bank-spread)

__device__ __forceinline__ unsigned pack_bf16(float a, float b) {
#if __has_builtin(__builtin_amdgcn_cvt_pk_bf16_f32)
  typedef __attribute__((ext_vector_type(2))) __bf16 v2bf;
  v2bf r = __builtin_amdgcn_cvt_pk_bf16_f32(a, b);   // dst.l = a, dst.h = b
  return __builtin_bit_cast(unsigned, r);
#else
  // Truncating f32->bf16 pack in ONE v_perm_b32: result = {hi16(b), hi16(a)}.
  // sel 0x07060302: b0<-S1.b2, b1<-S1.b3, b2<-S0.b2, b3<-S0.b3  (S0=b, S1=a)
  return __builtin_amdgcn_perm(__float_as_uint(b), __float_as_uint(a), 0x07060302u);
#endif
}

__device__ __forceinline__ unsigned short bf16_1(float a) {
  return (unsigned short)(__float_as_uint(a) >> 16);   // truncating f32->bf16
}

__global__ __launch_bounds__(BLOCKT) void ccam_fused(
    const float* __restrict__ x,
    const float* __restrict__ martx,
    const float* __restrict__ aphal,
    const float* __restrict__ gamma,
    float* __restrict__ out)
{
  // Double-buffered martx chunk, stored transposed (column-major over k) in bf16.
  __shared__ __align__(16) __bf16 ldsB[2][KD * BSTRIDE];          // 10240 B
  __shared__ __align__(16) __bf16 ldsAtt[WAVES][16 * ASTRIDE];    // 18432 B

  const int tid  = threadIdx.x;
  const int wave = tid >> 5;
  const int lane = tid & 31;
  const int lp   = lane & 15;     // position within 16-lane half
  const int half = lane >> 4;     // 0 or 1

  const int r0 = (blockIdx.x * WAVES + wave) * 16;   // first of this wave's 16 rows

  const float a_s = aphal[0];
  const float g_s = gamma[0];

  v8f acc[4];
  #pragma unroll
  for (int t = 0; t < 4; ++t) acc[t] = v8f{};

  // ---------------- Phase 1: energy = x-tile @ martx ----------------
  // A layout (16-bit 16x32): lane lp holds row M=lp; half selects k sub-runs:
  //   run1 = k in [8*half, 8*half+8), run2 = k in [16+8*half, 16+8*half+8)
  const float* xrow = x + (size_t)(r0 + lp) * NDIM;

  // Staging assignment: thread -> column c = tid&63, k-run kbase = (tid>>6)*8.
  const int st_c     = tid & 63;
  const int st_kbase = (tid >> 6) * 8;

  for (int it = 0; it < NITERS; ++it) {
    const int n0 = it * NCHUNK;
    __bf16* buf = ldsB[it & 1];

    // Cooperative stage: martx[n0..n0+31][0..63] -> LDS transposed bf16 [c*BSTRIDE + k]
    // 8 coalesced global_load_b32 + 4 v_perm + 1 ds_store_b128 per thread.
    {
      const float* src = martx + (size_t)(n0 + st_kbase) * KD + st_c;
      float f0 = src[0 * KD], f1 = src[1 * KD], f2 = src[2 * KD], f3 = src[3 * KD];
      float f4 = src[4 * KD], f5 = src[5 * KD], f6 = src[6 * KD], f7 = src[7 * KD];
      uint4 w;
      w.x = pack_bf16(f0, f1);
      w.y = pack_bf16(f2, f3);
      w.z = pack_bf16(f4, f5);
      w.w = pack_bf16(f6, f7);
      *(uint4*)&buf[st_c * BSTRIDE + st_kbase] = w;
    }

    if (it + 4 < NITERS)
      __builtin_prefetch(xrow + n0 + 4 * NCHUNK, 0, 0);

    __syncthreads();   // single barrier per iter; double buffer makes this safe

    // A fragment: 4x global_load_b128 from this lane's row + 8 v_perm packs
    BF16x16 af;
    {
      const float4* p1 = (const float4*)(xrow + n0 + half * 8);
      const float4* p2 = (const float4*)(xrow + n0 + 16 + half * 8);
      float4 r10 = p1[0], r11 = p1[1];
      float4 r20 = p2[0], r21 = p2[1];
      af.u[0] = pack_bf16(r10.x, r10.y);
      af.u[1] = pack_bf16(r10.z, r10.w);
      af.u[2] = pack_bf16(r11.x, r11.y);
      af.u[3] = pack_bf16(r11.z, r11.w);
      af.u[4] = pack_bf16(r20.x, r20.y);
      af.u[5] = pack_bf16(r20.z, r20.w);
      af.u[6] = pack_bf16(r21.x, r21.y);
      af.u[7] = pack_bf16(r21.z, r21.w);
    }

    // B frags (conflict-free ds_load_b128) + 4 accumulating WMMAs.
    uint4 bl[8];
    #pragma unroll
    for (int t = 0; t < 4; ++t) {
      const __bf16* col = &buf[(t * 16 + lp) * BSTRIDE + half * 8];
      bl[2 * t]     = *(const uint4*)col;          // k = 8h .. 8h+7
      bl[2 * t + 1] = *(const uint4*)(col + 16);   // k = 16+8h .. 16+8h+7
    }
    #pragma unroll
    for (int t = 0; t < 4; ++t) {
      BF16x16 bf_;
      bf_.u[0] = bl[2 * t].x;     bf_.u[1] = bl[2 * t].y;
      bf_.u[2] = bl[2 * t].z;     bf_.u[3] = bl[2 * t].w;
      bf_.u[4] = bl[2 * t + 1].x; bf_.u[5] = bl[2 * t + 1].y;
      bf_.u[6] = bl[2 * t + 1].z; bf_.u[7] = bl[2 * t + 1].w;
      acc[t] = __builtin_amdgcn_wmma_f32_16x16x32_bf16(
          false, af.v, false, bf_.v, (short)0, acc[t], false, false);
    }
  }

  // ---------------- Phase 2a: softmax over 64 per row ----------------
  // C/D layout: lane holds (row = v + 8*half, col = lp) for v = 0..7, per tile t.
  float p[4][8];
  #pragma unroll
  for (int v = 0; v < 8; ++v) {
    float e0 = acc[0][v], e1 = acc[1][v], e2 = acc[2][v], e3 = acc[3][v];
    // one combined pass: row max (m) and row min (mn)
    float m  = fmaxf(fmaxf(e0, e1), fmaxf(e2, e3));
    float mn = fminf(fminf(e0, e1), fminf(e2, e3));
    #pragma unroll
    for (int off = 1; off < 16; off <<= 1) {
      m  = fmaxf(m,  __shfl_xor(m,  off, 16));
      mn = fminf(mn, __shfl_xor(mn, off, 16));
    }
    // z = a*(m - e) has range endpoints at e=m (z=0) and e=mn (z=a*(m-mn));
    // max over the row is fmax of the two (handles either sign of a).
    float zm = fmaxf(a_s * (m - mn), 0.0f);
    float q0 = __expf(a_s * (m - e0) - zm);
    float q1 = __expf(a_s * (m - e1) - zm);
    float q2 = __expf(a_s * (m - e2) - zm);
    float q3 = __expf(a_s * (m - e3) - zm);
    float s = q0 + q1 + q2 + q3;
    #pragma unroll
    for (int off = 1; off < 16; off <<= 1)
      s += __shfl_xor(s, off, 16);
    float inv = 1.0f / s;
    p[0][v] = q0 * inv; p[1][v] = q1 * inv;
    p[2][v] = q2 * inv; p[3][v] = q3 * inv;
  }

  // Re-layout attention (C-layout -> A-layout) through padded LDS, as bf16.
  __bf16* attw = ldsAtt[wave];
  #pragma unroll
  for (int t = 0; t < 4; ++t)
    #pragma unroll
    for (int v = 0; v < 8; ++v)
      attw[(v + half * 8) * ASTRIDE + t * 16 + lp] =
          __builtin_bit_cast(__bf16, bf16_1(p[t][v]));
  asm volatile("s_wait_dscnt 0" ::: "memory");  // within-wave cross-lane LDS visibility

  // Attention A-frags (16x64 bf16 -> two 16x32 frags), invariant over n-loop.
  BF16x16 aF0, aF1;
  {
    const __bf16* ar = attw + lp * ASTRIDE + half * 8;
    uint4 t0 = *(const uint4*)(ar);        // k = 8h..
    uint4 t1 = *(const uint4*)(ar + 16);   // k = 16+8h..
    uint4 t2 = *(const uint4*)(ar + 32);   // k = 32+8h..
    uint4 t3 = *(const uint4*)(ar + 48);   // k = 48+8h..
    aF0.u[0] = t0.x; aF0.u[1] = t0.y; aF0.u[2] = t0.z; aF0.u[3] = t0.w;
    aF0.u[4] = t1.x; aF0.u[5] = t1.y; aF0.u[6] = t1.z; aF0.u[7] = t1.w;
    aF1.u[0] = t2.x; aF1.u[1] = t2.y; aF1.u[2] = t2.z; aF1.u[3] = t2.w;
    aF1.u[4] = t3.x; aF1.u[5] = t3.y; aF1.u[6] = t3.z; aF1.u[7] = t3.w;
  }

  // ---------------- Phase 2b: out = gamma * (att @ martx^T) + x ----------------
  // B[k, c] = martx[n0 + c][k]  (k contiguous along martx rows -> b128 loads)
  #pragma unroll 2
  for (int nt = 0; nt < NDIM / 16; ++nt) {
    const int n0 = nt * 16;
    const float* mp = martx + (size_t)(n0 + lp) * KD + half * 8;

    BF16x16 b0, b1;
    {
      float4 u0 = ((const float4*)mp)[0];
      float4 u1 = ((const float4*)mp)[1];
      float4 u2 = ((const float4*)(mp + 16))[0];
      float4 u3 = ((const float4*)(mp + 16))[1];
      b0.u[0] = pack_bf16(u0.x, u0.y); b0.u[1] = pack_bf16(u0.z, u0.w);
      b0.u[2] = pack_bf16(u1.x, u1.y); b0.u[3] = pack_bf16(u1.z, u1.w);
      b0.u[4] = pack_bf16(u2.x, u2.y); b0.u[5] = pack_bf16(u2.z, u2.w);
      b0.u[6] = pack_bf16(u3.x, u3.y); b0.u[7] = pack_bf16(u3.z, u3.w);
      float4 w0 = ((const float4*)(mp + 32))[0];
      float4 w1 = ((const float4*)(mp + 32))[1];
      float4 w2 = ((const float4*)(mp + 48))[0];
      float4 w3 = ((const float4*)(mp + 48))[1];
      b1.u[0] = pack_bf16(w0.x, w0.y); b1.u[1] = pack_bf16(w0.z, w0.w);
      b1.u[2] = pack_bf16(w1.x, w1.y); b1.u[3] = pack_bf16(w1.z, w1.w);
      b1.u[4] = pack_bf16(w2.x, w2.y); b1.u[5] = pack_bf16(w2.z, w2.w);
      b1.u[6] = pack_bf16(w3.x, w3.y); b1.u[7] = pack_bf16(w3.z, w3.w);
    }

    v8f o = v8f{};
    o = __builtin_amdgcn_wmma_f32_16x16x32_bf16(false, aF0.v, false, b0.v, (short)0, o, false, false);
    o = __builtin_amdgcn_wmma_f32_16x16x32_bf16(false, aF1.v, false, b1.v, (short)0, o, false, false);

    // Residual + nontemporal store; lanes 0..15 cover 64B per row -> coalesced
    #pragma unroll
    for (int v = 0; v < 8; ++v) {
      size_t idx = (size_t)(r0 + v + half * 8) * NDIM + (size_t)(n0 + lp);
      float r = __builtin_fmaf(g_s, o[v], x[idx]);
      __builtin_nontemporal_store(r, &out[idx]);
    }
  }
}

extern "C" void kernel_launch(void* const* d_in, const int* in_sizes, int n_in,
                              void* d_out, int out_size, void* d_ws, size_t ws_size,
                              hipStream_t stream) {
  (void)in_sizes; (void)n_in; (void)out_size; (void)d_ws; (void)ws_size;
  const float* x     = (const float*)d_in[0];
  const float* martx = (const float*)d_in[1];
  const float* aphal = (const float*)d_in[2];
  const float* gamma = (const float*)d_in[3];
  float* out = (float*)d_out;

  const int rows = 16 * 1024;                 // B*C
  dim3 grid(rows / (WAVES * 16));             // 128 workgroups
  dim3 block(BLOCKT);                         // 8 waves (wave32)
  ccam_fused<<<grid, block, 0, stream>>>(x, martx, aphal, gamma, out);
}
